// RNN_13469017441119
// MI455X (gfx1250) — matrix operand order repaired and checked by reading, not compiled, not run on hardware
//
#include <hip/hip_runtime.h>
#include <hip/hip_bf16.h>
#include <math.h>

// Problem constants (match reference): B=64, T=128, I=2048, H=1024
#define B_DIM 64
#define T_DIM 128
#define I_DIM 2048
#define H_DIM 1024
#define BT_DIM (B_DIM * T_DIM)   // 8192

typedef __attribute__((ext_vector_type(16))) _Float16 v16h;
typedef __attribute__((ext_vector_type(8)))  _Float16 h8;
typedef __attribute__((ext_vector_type(8)))  float    v8f;
typedef __attribute__((ext_vector_type(4)))  float    f32x4;

__device__ __forceinline__ v8f wmma_f16(v16h a, v16h b, v8f c) {
    // D(16x16,f32) = A(16x32,f16) x B(32x16,f16) + C(16x16,f32)
    return __builtin_amdgcn_wmma_f32_16x16x32_f16(
        /*neg_a=*/false, a, /*neg_b=*/false, b,
        /*c_mod=*/(short)0, c, /*reuse_a=*/false, /*reuse_b=*/false);
}

// ===========================================================================
// f16-direct fragment loaders (main path: operands pre-converted to f16)
// ===========================================================================

// A fragment (16x32 f16): lane (m = lane%16) holds K = kc+16*half..+15 (elems
// 0..15, contiguous). p = arow + kc + 16*half.
__device__ __forceinline__ v16h load_a_frag16(const _Float16* __restrict__ p) {
    h8 lo = *reinterpret_cast<const h8*>(p);        // 16B global_load_b128
    h8 hi = *reinterpret_cast<const h8*>(p + 8);    // 16B global_load_b128
    v16h r;
#pragma unroll
    for (int i = 0; i < 8; ++i) { r[i] = lo[i]; r[8 + i] = hi[i]; }
    return r;
}

// B fragment (32x16 f16): lane (n = lane%16) holds W[n][kc+16*half .. +15] —
// 16 contiguous halves (32B, 32B-aligned) of a row of the (N,K) weight.
__device__ __forceinline__ v16h load_b_frag16(const _Float16* __restrict__ p) {
    return *reinterpret_cast<const v16h*>(p);       // 2x global_load_b128
}

__device__ __forceinline__ void gemm_16x64_f16(
    const _Float16* __restrict__ arow,
    const _Float16* __restrict__ w0, const _Float16* __restrict__ w1,
    const _Float16* __restrict__ w2, const _Float16* __restrict__ w3,
    int K, int half, v8f c[4])
{
    const _Float16* ap = arow + 16 * half;
    for (int kc = 0; kc < K; kc += 32) {
        v16h a  = load_a_frag16(ap + kc);
        v16h b0 = load_b_frag16(w0 + kc);
        v16h b1 = load_b_frag16(w1 + kc);
        v16h b2 = load_b_frag16(w2 + kc);
        v16h b3 = load_b_frag16(w3 + kc);
        c[0] = wmma_f16(a, b0, c[0]);
        c[1] = wmma_f16(a, b1, c[1]);
        c[2] = wmma_f16(a, b2, c[2]);
        c[3] = wmma_f16(a, b3, c[3]);
    }
}

// ===========================================================================
// f32-load + inline-convert fragment loaders (fallback paths)
// ===========================================================================

__device__ __forceinline__ v16h load_b_frag(const float* __restrict__ p) {
    f32x4 f[4];
#pragma unroll
    for (int q = 0; q < 4; ++q) f[q] = reinterpret_cast<const f32x4*>(p)[q];
    v16h r;
#pragma unroll
    for (int q = 0; q < 4; ++q)
#pragma unroll
        for (int i = 0; i < 4; ++i) r[4 * q + i] = (_Float16)f[q][i];
    return r;
}

__device__ __forceinline__ v16h load_a_frag(const float* __restrict__ p) {
    f32x4 f[4];
    f[0] = reinterpret_cast<const f32x4*>(p)[0];
    f[1] = reinterpret_cast<const f32x4*>(p)[1];
    f[2] = reinterpret_cast<const f32x4*>(p + 8)[0];
    f[3] = reinterpret_cast<const f32x4*>(p + 8)[1];
    v16h r;
#pragma unroll
    for (int q = 0; q < 4; ++q)
#pragma unroll
        for (int i = 0; i < 4; ++i) r[4 * q + i] = (_Float16)f[q][i];
    return r;
}

__device__ __forceinline__ void gemm_16x64_f32cvt(
    const float* __restrict__ arow,
    const float* __restrict__ w0, const float* __restrict__ w1,
    const float* __restrict__ w2, const float* __restrict__ w3,
    int K, int half, v8f c[4])
{
    const float* ap = arow + 16 * half;
    for (int kc = 0; kc < K; kc += 32) {
        v16h a  = load_a_frag(ap + kc);
        v16h b0 = load_b_frag(w0 + kc);
        v16h b1 = load_b_frag(w1 + kc);
        v16h b2 = load_b_frag(w2 + kc);
        v16h b3 = load_b_frag(w3 + kc);
        c[0] = wmma_f16(a, b0, c[0]);
        c[1] = wmma_f16(a, b1, c[1]);
        c[2] = wmma_f16(a, b2, c[2]);
        c[3] = wmma_f16(a, b3, c[3]);
    }
}

// ===========================================================================
// Prologue: bulk f32 -> f16 conversion (8 elems/thread, b128 in / b128 out)
// ===========================================================================
__global__ void __launch_bounds__(256) cvt_f32_f16_kernel(
    const float* __restrict__ src, _Float16* __restrict__ dst, int n8)
{
    const int i = blockIdx.x * blockDim.x + threadIdx.x;
    if (i < n8) {
        f32x4 a = reinterpret_cast<const f32x4*>(src)[2 * (size_t)i];
        f32x4 b = reinterpret_cast<const f32x4*>(src)[2 * (size_t)i + 1];
        h8 r;
#pragma unroll
        for (int j = 0; j < 4; ++j) { r[j] = (_Float16)a[j]; r[4 + j] = (_Float16)b[j]; }
        reinterpret_cast<h8*>(dst)[i] = r;
    }
}

// ===========================================================================
// Phase 1 (main): xp[8192,1024] = x16 @ Wih16^T + (b_ih + b_hh)
// Block = 256 (8 waves) -> 128(M) x 64(N); grid = (64, 16).
// ===========================================================================
__global__ void __launch_bounds__(256) xp_gemm_f16_kernel(
    const _Float16* __restrict__ x16, const _Float16* __restrict__ Wih16,
    const float* __restrict__ bih, const float* __restrict__ bhh,
    float* __restrict__ xp)
{
    const int lane = threadIdx.x & 31;
    const int wave = threadIdx.x >> 5;
    const int half = lane >> 4;
    const int lm   = lane & 15;
    const int m0 = blockIdx.x * 128 + wave * 16;
    const int n0 = blockIdx.y * 64;

    const _Float16* arow = x16 + (size_t)(m0 + lm) * I_DIM;
    const _Float16* w0 = Wih16 + (size_t)(n0 +  0 + lm) * I_DIM + 16 * half;
    const _Float16* w1 = Wih16 + (size_t)(n0 + 16 + lm) * I_DIM + 16 * half;
    const _Float16* w2 = Wih16 + (size_t)(n0 + 32 + lm) * I_DIM + 16 * half;
    const _Float16* w3 = Wih16 + (size_t)(n0 + 48 + lm) * I_DIM + 16 * half;

    v8f c[4];
    const v8f z = {};
#pragma unroll
    for (int j = 0; j < 4; ++j) c[j] = z;

    gemm_16x64_f16(arow, w0, w1, w2, w3, I_DIM, half, c);

#pragma unroll
    for (int j = 0; j < 4; ++j) {
        const int n = n0 + j * 16 + lm;
        const float bias = bih[n] + bhh[n];
#pragma unroll
        for (int v = 0; v < 8; ++v) {
            const int m = m0 + 8 * half + v;
            xp[(size_t)m * H_DIM + n] = c[j][v] + bias;
        }
    }
}

// ===========================================================================
// Phase 2 (main): h_out16 = f16(tanh(xp[:,t,:] + h_in16 @ Whh16^T)),
// optionally also store f32 result (last step -> d_out). Grid = 8 x 256.
// ===========================================================================
__global__ void __launch_bounds__(256) rnn_step_f16_kernel(
    const float* __restrict__ xp, const _Float16* __restrict__ h_in,
    const _Float16* __restrict__ Whh16, _Float16* __restrict__ h_out,
    float* __restrict__ out32, int t, int first)
{
    const int lane = threadIdx.x & 31;
    const int wave = threadIdx.x >> 5;
    const int half = lane >> 4;
    const int lm   = lane & 15;
    const int m0 = (wave & 3) * 16;
    const int n0 = blockIdx.x * 128 + (wave >> 2) * 64;

    v8f c[4];
    const v8f z = {};
#pragma unroll
    for (int j = 0; j < 4; ++j) c[j] = z;

    if (!first) {
        const _Float16* arow = h_in + (size_t)(m0 + lm) * H_DIM;
        const _Float16* w0 = Whh16 + (size_t)(n0 +  0 + lm) * H_DIM + 16 * half;
        const _Float16* w1 = Whh16 + (size_t)(n0 + 16 + lm) * H_DIM + 16 * half;
        const _Float16* w2 = Whh16 + (size_t)(n0 + 32 + lm) * H_DIM + 16 * half;
        const _Float16* w3 = Whh16 + (size_t)(n0 + 48 + lm) * H_DIM + 16 * half;
        gemm_16x64_f16(arow, w0, w1, w2, w3, H_DIM, half, c);
    }

#pragma unroll
    for (int j = 0; j < 4; ++j) {
        const int n = n0 + j * 16 + lm;
#pragma unroll
        for (int v = 0; v < 8; ++v) {
            const int m = m0 + 8 * half + v;
            const float val = tanhf(c[j][v] + xp[((size_t)m * T_DIM + t) * H_DIM + n]);
            h_out[(size_t)m * H_DIM + n] = (_Float16)val;
            if (out32) out32[(size_t)m * H_DIM + n] = val;
        }
    }
}

// ===========================================================================
// Fallback tier 2: round-1 kernels (f32 loads, inline convert), xp in ws
// ===========================================================================
__global__ void __launch_bounds__(256) xp_gemm_kernel(
    const float* __restrict__ x, const float* __restrict__ Wih,
    const float* __restrict__ bih, const float* __restrict__ bhh,
    float* __restrict__ xp)
{
    const int lane = threadIdx.x & 31;
    const int wave = threadIdx.x >> 5;
    const int half = lane >> 4;
    const int lm   = lane & 15;
    const int m0 = blockIdx.x * 128 + wave * 16;
    const int n0 = blockIdx.y * 64;

    const float* arow = x + (size_t)(m0 + lm) * I_DIM;
    const float* w0 = Wih + (size_t)(n0 +  0 + lm) * I_DIM + 16 * half;
    const float* w1 = Wih + (size_t)(n0 + 16 + lm) * I_DIM + 16 * half;
    const float* w2 = Wih + (size_t)(n0 + 32 + lm) * I_DIM + 16 * half;
    const float* w3 = Wih + (size_t)(n0 + 48 + lm) * I_DIM + 16 * half;

    v8f c[4];
    const v8f z = {};
#pragma unroll
    for (int j = 0; j < 4; ++j) c[j] = z;

    gemm_16x64_f32cvt(arow, w0, w1, w2, w3, I_DIM, half, c);

#pragma unroll
    for (int j = 0; j < 4; ++j) {
        const int n = n0 + j * 16 + lm;
        const float bias = bih[n] + bhh[n];
#pragma unroll
        for (int v = 0; v < 8; ++v) {
            const int m = m0 + 8 * half + v;
            xp[(size_t)m * H_DIM + n] = c[j][v] + bias;
        }
    }
}

__global__ void __launch_bounds__(256) rnn_step_kernel(
    const float* __restrict__ xp, const float* __restrict__ h_in,
    const float* __restrict__ Whh, float* __restrict__ h_out,
    int t, int first)
{
    const int lane = threadIdx.x & 31;
    const int wave = threadIdx.x >> 5;
    const int half = lane >> 4;
    const int lm   = lane & 15;
    const int m0 = (wave & 3) * 16;
    const int n0 = blockIdx.x * 128 + (wave >> 2) * 64;

    v8f c[4];
    const v8f z = {};
#pragma unroll
    for (int j = 0; j < 4; ++j) c[j] = z;

    if (!first) {
        const float* arow = h_in + (size_t)(m0 + lm) * H_DIM;
        const float* w0 = Whh + (size_t)(n0 +  0 + lm) * H_DIM + 16 * half;
        const float* w1 = Whh + (size_t)(n0 + 16 + lm) * H_DIM + 16 * half;
        const float* w2 = Whh + (size_t)(n0 + 32 + lm) * H_DIM + 16 * half;
        const float* w3 = Whh + (size_t)(n0 + 48 + lm) * H_DIM + 16 * half;
        gemm_16x64_f32cvt(arow, w0, w1, w2, w3, H_DIM, half, c);
    }

#pragma unroll
    for (int j = 0; j < 4; ++j) {
        const int n = n0 + j * 16 + lm;
#pragma unroll
        for (int v = 0; v < 8; ++v) {
            const int m = m0 + 8 * half + v;
            const float val = c[j][v] + xp[((size_t)m * T_DIM + t) * H_DIM + n];
            h_out[(size_t)m * H_DIM + n] = tanhf(val);
        }
    }
}

// ===========================================================================
// Fallback tier 3: fused per-step (minimal ws)
// ===========================================================================
__global__ void __launch_bounds__(256) rnn_step_fused_kernel(
    const float* __restrict__ x, const float* __restrict__ Wih,
    const float* __restrict__ bih, const float* __restrict__ bhh,
    const float* __restrict__ h_in, const float* __restrict__ Whh,
    float* __restrict__ h_out, int t, int first)
{
    const int lane = threadIdx.x & 31;
    const int wave = threadIdx.x >> 5;
    const int half = lane >> 4;
    const int lm   = lane & 15;
    const int m0 = (wave & 3) * 16;
    const int n0 = blockIdx.x * 128 + (wave >> 2) * 64;

    v8f c[4];
    const v8f z = {};
#pragma unroll
    for (int j = 0; j < 4; ++j) c[j] = z;

    {
        const float* arow = x + ((size_t)(m0 + lm) * T_DIM + t) * I_DIM;
        const float* w0 = Wih + (size_t)(n0 +  0 + lm) * I_DIM + 16 * half;
        const float* w1 = Wih + (size_t)(n0 + 16 + lm) * I_DIM + 16 * half;
        const float* w2 = Wih + (size_t)(n0 + 32 + lm) * I_DIM + 16 * half;
        const float* w3 = Wih + (size_t)(n0 + 48 + lm) * I_DIM + 16 * half;
        gemm_16x64_f32cvt(arow, w0, w1, w2, w3, I_DIM, half, c);
    }
    if (!first) {
        const float* arow = h_in + (size_t)(m0 + lm) * H_DIM;
        const float* w0 = Whh + (size_t)(n0 +  0 + lm) * H_DIM + 16 * half;
        const float* w1 = Whh + (size_t)(n0 + 16 + lm) * H_DIM + 16 * half;
        const float* w2 = Whh + (size_t)(n0 + 32 + lm) * H_DIM + 16 * half;
        const float* w3 = Whh + (size_t)(n0 + 48 + lm) * H_DIM + 16 * half;
        gemm_16x64_f32cvt(arow, w0, w1, w2, w3, H_DIM, half, c);
    }

#pragma unroll
    for (int j = 0; j < 4; ++j) {
        const int n = n0 + j * 16 + lm;
        const float bias = bih[n] + bhh[n];
#pragma unroll
        for (int v = 0; v < 8; ++v) {
            const int m = m0 + 8 * half + v;
            h_out[(size_t)m * H_DIM + n] = tanhf(c[j][v] + bias);
        }
    }
}

extern "C" void kernel_launch(void* const* d_in, const int* in_sizes, int n_in,
                              void* d_out, int out_size, void* d_ws, size_t ws_size,
                              hipStream_t stream) {
    (void)in_sizes; (void)n_in; (void)out_size;
    const float* x   = (const float*)d_in[0];   // (B, T, I)
    const float* Wih = (const float*)d_in[1];   // (H, I)
    const float* Whh = (const float*)d_in[2];   // (H, H)
    const float* bih = (const float*)d_in[3];   // (H,)
    const float* bhh = (const float*)d_in[4];   // (H,)
    float* out = (float*)d_out;                 // (B, H)

    const size_t x_e   = (size_t)BT_DIM * I_DIM;   // 16.78M
    const size_t wih_e = (size_t)H_DIM * I_DIM;    // 2M
    const size_t whh_e = (size_t)H_DIM * H_DIM;    // 1M
    const size_t xp_e  = (size_t)BT_DIM * H_DIM;   // 8M
    const size_t h_e   = (size_t)B_DIM * H_DIM;    // 64K

    const size_t need1 = xp_e * 4 + (x_e + wih_e + whh_e + 2 * h_e) * 2; // ~72 MB
    const size_t need2 = (xp_e + 2 * h_e) * 4;                           // ~34 MB

    if (ws_size >= need1) {
        // ---- Tier 1: pre-convert everything to f16; pure-f16 WMMA loops ----
        float*     xp    = (float*)d_ws;
        _Float16*  x16   = (_Float16*)(xp + xp_e);
        _Float16*  Wih16 = x16 + x_e;
        _Float16*  Whh16 = Wih16 + wih_e;
        _Float16*  h16a  = Whh16 + whh_e;
        _Float16*  h16b  = h16a + h_e;

        cvt_f32_f16_kernel<<<(int)((x_e / 8 + 255) / 256), 256, 0, stream>>>(x, x16, (int)(x_e / 8));
        cvt_f32_f16_kernel<<<(int)((wih_e / 8 + 255) / 256), 256, 0, stream>>>(Wih, Wih16, (int)(wih_e / 8));
        cvt_f32_f16_kernel<<<(int)((whh_e / 8 + 255) / 256), 256, 0, stream>>>(Whh, Whh16, (int)(whh_e / 8));

        xp_gemm_f16_kernel<<<dim3(64, 16), 256, 0, stream>>>(x16, Wih16, bih, bhh, xp);

        const _Float16* hin = h16a;   // unused when first==1
        for (int t = 0; t < T_DIM; ++t) {
            _Float16* hout = (t & 1) ? h16b : h16a;
            float* out32 = (t == T_DIM - 1) ? out : nullptr;
            rnn_step_f16_kernel<<<8, 256, 0, stream>>>(xp, hin, Whh16, hout, out32, t, t == 0);
            hin = hout;
        }
    } else if (ws_size >= need2) {
        // ---- Tier 2: xp precompute, inline f32->f16 conversion ----
        float* xp = (float*)d_ws;
        float* hbuf0 = xp + xp_e;
        float* hbuf1 = hbuf0 + h_e;

        xp_gemm_kernel<<<dim3(64, 16), 256, 0, stream>>>(x, Wih, bih, bhh, xp);

        const float* hin = hbuf0;
        for (int t = 0; t < T_DIM; ++t) {
            float* hout = (t == T_DIM - 1) ? out : ((t & 1) ? hbuf1 : hbuf0);
            rnn_step_kernel<<<8, 256, 0, stream>>>(xp, hin, Whh, hout, t, t == 0);
            hin = hout;
        }
    } else {
        // ---- Tier 3: fused per-step, minimal ws ----
        float* hbuf0 = (float*)d_ws;
        float* hbuf1 = hbuf0 + h_e;
        const float* hin = hbuf0;
        for (int t = 0; t < T_DIM; ++t) {
            float* hout = (t == T_DIM - 1) ? out : ((t & 1) ? hbuf1 : hbuf0);
            rnn_step_fused_kernel<<<8, 256, 0, stream>>>(
                x, Wih, bih, bhh, hin, Whh, hout, t, t == 0);
            hin = hout;
        }
    }
}
